// SphericalMultiHeadAttention_12275016532044
// MI455X (gfx1250) — matrix operand order repaired and checked by reading, not compiled, not run on hardware
//
#include <hip/hip_runtime.h>

// ---------------------------------------------------------------------------
// SphericalMultiHeadAttention on MI455X (gfx1250, wave32, WMMA).
//
// Compact-spectral formulation (everything off the 1024 lm_idx positions is
// zero end-to-end). Four big GEMMs on V_WMMA_F32_16X16X32_F16 (f16 in, f32
// accumulate). B operands staged K-major so fragments are ds_load_b128 pairs.
// Global->LDS staging uses CDNA5 async-to-LDS DMA (ASYNCcnt) with a
// double-buffered pipeline so DMA overlaps the WMMA pipe.
// ---------------------------------------------------------------------------

#define LBAND 32
#define MPHI  63
#define SSP   (LBAND * MPHI)   // 2016 spatial samples
#define NLM   1024             // L*L compact spectral size
#define NC2   2048             // 2*NLM (complex interleaved) == padded N
#define BB    32
#define CCH   64               // IN_C == OUT_C
#define RQK   2048             // B * H * D rows
#define WROWS 5632             // 512*3 (q,k,v) + 4096 (o) weight rows

typedef __attribute__((ext_vector_type(16))) _Float16 v16h;
typedef __attribute__((ext_vector_type(8)))  float    v8f;
typedef __attribute__((ext_vector_type(4)))  int      v4i;

union FragU {
  uint4 q[2];
  v16h  h;
};

// ---- CDNA5 async global->LDS copy (ASYNCcnt) ------------------------------

#if __has_builtin(__builtin_amdgcn_global_load_async_to_lds_b128)
#define USE_ASYNC_LDS 1
typedef __attribute__((address_space(1))) v4i* gv4i_ptr;
typedef __attribute__((address_space(3))) v4i* lv4i_ptr;
#else
#define USE_ASYNC_LDS 0
#endif

__device__ __forceinline__ void lds_cp16(_Float16* l, const _Float16* g) {
#if USE_ASYNC_LDS
  __builtin_amdgcn_global_load_async_to_lds_b128(
      (gv4i_ptr)const_cast<_Float16*>(g), (lv4i_ptr)l, 0, 0);
#else
  *(uint4*)l = *(const uint4*)g;
#endif
}

__device__ __forceinline__ void async_drain() {
#if USE_ASYNC_LDS
#if __has_builtin(__builtin_amdgcn_s_wait_asynccnt)
  __builtin_amdgcn_s_wait_asynccnt(0);
#else
  asm volatile("s_wait_asynccnt 0" ::: "memory");
#endif
#endif
}

// ---- WMMA fragment helpers (contiguous k order, same in A and B, wave32) ---------

// A 16x32 f16, LDS row-major [rows][ld]: lanes 0-15 row M=lane, K
// kb..kb+15 contiguous with kb=0; lanes 16-31 kb=16. Two b128 loads.
__device__ __forceinline__ v16h frag_a(const _Float16* s, int row0, int k0,
                                       int ld, int lane) {
  int r  = row0 + (lane & 15);
  int kb = k0 + (lane & 16);
  const uint4* p = (const uint4*)(s + r * ld + kb);
  FragU f;
  f.q[0] = p[0];   // K kb .. kb+7
  f.q[1] = p[1];   // K kb+8 .. kb+15
  return f.h;
}

// B 32x16 f16 from K-major (transposed) LDS [cols][ld]: lane column N,
// K kb..kb+15 contiguous (kb = 0 / 16 by half-wave). Two b128 loads.
__device__ __forceinline__ v16h frag_bt(const _Float16* s, int col0, int k0,
                                        int ld, int lane) {
  int n  = col0 + (lane & 15);
  int kb = k0 + (lane & 16);
  const uint4* p = (const uint4*)(s + n * ld + kb);
  FragU f;
  f.q[0] = p[0];
  f.q[1] = p[1];
  return f.h;
}

#define WMMA_F32_F16(a, b, c) \
  __builtin_amdgcn_wmma_f32_16x16x32_f16(false, (a), false, (b), (short)0, (c), false, false)

// ---- small utility kernels -------------------------------------------------

__global__ void zero_f32_kernel(float* p, int n) {
  int i = blockIdx.x * 256 + threadIdx.x;
  if (i < n) p[i] = 0.0f;
}

__global__ void cvt_f16_kernel(const float* __restrict__ src,
                               _Float16* __restrict__ dst, int n) {
  int i = blockIdx.x * 256 + threadIdx.x;
  if (i < n) dst[i] = (_Float16)src[i];
}

// Ypinv [NLM, SSP] complex -> YpTT f16 [2048][SSP] (K-major B^T):
// row 2k+j holds {re,im}(Ypinv[k][:]).
__global__ void pack_yptt_kernel(const float* __restrict__ yp,
                                 _Float16* __restrict__ out) {
  int gid = blockIdx.x * 256 + threadIdx.x;   // NC2*SSP threads
  int s   = gid % SSP;
  int row = gid / SSP;                        // 2k + j
  int k = row >> 1, j = row & 1;
  out[(size_t)row * SSP + s] = (_Float16)yp[((size_t)k * SSP + s) * 2 + j];
}

// Ymat [SSP, NLM] complex -> Ym2TT f16 [2048][2048] (K-major B^T):
// row s (<SSP): cols 0..1023 = Re(Ymat[s][k]), cols 1024..2047 = -Im(Ymat[s][k]);
// rows >= SSP zero (padding).
__global__ void pack_ym2tt_kernel(const float* __restrict__ ym,
                                  _Float16* __restrict__ out) {
  int gid = blockIdx.x * 256 + threadIdx.x;   // NC2*NC2 threads
  int kk = gid & (NC2 - 1);
  int s  = gid >> 11;
  float v = 0.0f;
  if (s < SSP) {
    if (kk < NLM) v = ym[((size_t)s * NLM + kk) * 2];
    else          v = -ym[((size_t)s * NLM + (kk - NLM)) * 2 + 1];
  }
  out[(size_t)s * NC2 + kk] = (_Float16)v;
}

// ---- generic WMMA GEMM: C[M,N] f32 = A[M,K] f16 @ (BT[N,K])^T f16 ---------
// 64x64 C-tile per block, 128 threads (4 waves), BK = 32, double-buffered
// LDS with async global->LDS DMA overlapping the WMMA pipe.

struct Tile {
  uint4 a0, a1, b0, b1;
};

__device__ __forceinline__ void tile_issue(Tile& tl,
                                           const _Float16* ga, const _Float16* gb,
                                           _Float16* la, _Float16* lb) {
#if USE_ASYNC_LDS
  (void)tl;
  lds_cp16(la, ga);
  lds_cp16(la + 8, ga + 8);
  lds_cp16(lb, gb);
  lds_cp16(lb + 8, gb + 8);
#else
  tl.a0 = *(const uint4*)ga;
  tl.a1 = *(const uint4*)(ga + 8);
  tl.b0 = *(const uint4*)gb;
  tl.b1 = *(const uint4*)(gb + 8);
  __builtin_prefetch(ga + 32, 0, 1);
  __builtin_prefetch(gb + 32, 0, 1);
#endif
}

__device__ __forceinline__ void tile_commit(const Tile& tl,
                                            _Float16* la, _Float16* lb) {
#if !USE_ASYNC_LDS
  *(uint4*)la       = tl.a0;
  *(uint4*)(la + 8) = tl.a1;
  *(uint4*)lb       = tl.b0;
  *(uint4*)(lb + 8) = tl.b1;
#endif
}

__global__ __launch_bounds__(128) void gemm_wmma_kernel(
    const _Float16* __restrict__ A, const _Float16* __restrict__ BT,
    float* __restrict__ C, int Mdim, int Ndim, int Kdim) {
  __shared__ __align__(16) _Float16 sA[2][64 * 32];
  __shared__ __align__(16) _Float16 sBT[2][64 * 32];
  const int bm = blockIdx.y * 64;
  const int bn = blockIdx.x * 64;
  const int t = threadIdx.x;
  const int wave = t >> 5;
  const int lane = t & 31;
  v8f acc[4] = {};

  const int r  = t >> 1;
  const int cb = (t & 1) * 16;
  const _Float16* Ag = A  + (size_t)(bm + r) * Kdim + cb;
  const _Float16* Bg = BT + (size_t)(bn + r) * Kdim + cb;
  _Float16* la[2] = { &sA[0][r * 32 + cb],  &sA[1][r * 32 + cb] };
  _Float16* lb[2] = { &sBT[0][r * 32 + cb], &sBT[1][r * 32 + cb] };

  Tile tl;
  tile_issue(tl, Ag, Bg, la[0], lb[0]);
  tile_commit(tl, la[0], lb[0]);
  async_drain();
  __syncthreads();

  int buf = 0;
  for (int k0 = 0; k0 < Kdim; k0 += 32) {
    const bool has_next = (k0 + 32) < Kdim;
    if (has_next)
      tile_issue(tl, Ag + k0 + 32, Bg + k0 + 32, la[buf ^ 1], lb[buf ^ 1]);

    v16h af = frag_a(&sA[buf][0], wave * 16, 0, 32, lane);
#pragma unroll
    for (int nt = 0; nt < 4; ++nt) {
      v16h bf = frag_bt(&sBT[buf][0], nt * 16, 0, 32, lane);
      acc[nt] = WMMA_F32_F16(af, bf, acc[nt]);
    }

    if (has_next) tile_commit(tl, la[buf ^ 1], lb[buf ^ 1]);
    async_drain();
    __syncthreads();
    buf ^= 1;
  }
#pragma unroll
  for (int nt = 0; nt < 4; ++nt) {
    int n  = bn + nt * 16 + (lane & 15);
    int r0 = bm + wave * 16 + ((lane & 16) >> 1);
#pragma unroll
    for (int j = 0; j < 8; ++j)
      C[(size_t)(r0 + j) * Ndim + n] = acc[nt][j];
  }
}

// ---- FQ/FK/FV (c-contraction D=8) + P = FQ * conj(FK) ----------------------
// Wspec rows: [0,512)=kq, [512,1024)=kk, [1024,1536)=kv, [1536,5632)=ko;
// row within a block = h*64 + c*8 + d; columns interleaved re/im (2048 f32).

__global__ void qkv_kernel(const float* __restrict__ f_in,
                           const float* __restrict__ Wspec,
                           const int* __restrict__ lm_idx,
                           _Float16* __restrict__ Ph,
                           float* __restrict__ FV) {
  int gid = blockIdx.x * 256 + threadIdx.x;   // RQK * NLM threads
  int k = gid & (NLM - 1);
  int r = gid >> 10;                          // b*64 + h*8 + d
  int b = r >> 6;
  int h = (r >> 3) & 7;
  int d = r & 7;
  int sp = lm_idx[k];
  const float* fb = f_in + (size_t)(b * 64 + h * 8) * SSP + sp;
  const float* wq = Wspec + (size_t)(h * 64 + d) * NC2 + 2 * k;
  float qr = 0, qi = 0, kr = 0, ki = 0, vr = 0, vi = 0;
#pragma unroll
  for (int c = 0; c < 8; ++c) {
    float f = fb[(size_t)c * SSP];
    const float* w = wq + (size_t)c * 8 * NC2;
    qr += f * w[0];               qi += f * w[1];
    kr += f * w[512 * NC2];       ki += f * w[512 * NC2 + 1];
    vr += f * w[1024 * NC2];      vi += f * w[1024 * NC2 + 1];
  }
  float pr = qr * kr + qi * ki;   // Re(FQ * conj(FK))
  float pi = qi * kr - qr * ki;   // Im(FQ * conj(FK))
  Ph[(size_t)r * NC2 + k]        = (_Float16)pr;
  Ph[(size_t)r * NC2 + NLM + k]  = (_Float16)pi;
  FV[(size_t)r * NC2 + 2 * k]     = vr;
  FV[(size_t)r * NC2 + 2 * k + 1] = vi;
}

// ---- row softmax over 2016 spatial samples; writes f16 A for next GEMM -----

__global__ __launch_bounds__(256) void softmax_kernel(
    const float* __restrict__ Asp, _Float16* __restrict__ Ah) {
  int r = blockIdx.x;
  const float* row = Asp + (size_t)r * NC2;
  __shared__ float red[8];
  float m = -1e30f;
  for (int i = threadIdx.x; i < SSP; i += 256) m = fmaxf(m, row[i]);
#pragma unroll
  for (int o = 16; o > 0; o >>= 1) m = fmaxf(m, __shfl_xor(m, o, 32));
  if ((threadIdx.x & 31) == 0) red[threadIdx.x >> 5] = m;
  __syncthreads();
  float mm = red[0];
#pragma unroll
  for (int w = 1; w < 8; ++w) mm = fmaxf(mm, red[w]);
  __syncthreads();
  float s = 0.0f;
  for (int i = threadIdx.x; i < SSP; i += 256) s += __expf(row[i] - mm);
#pragma unroll
  for (int o = 16; o > 0; o >>= 1) s += __shfl_xor(s, o, 32);
  if ((threadIdx.x & 31) == 0) red[threadIdx.x >> 5] = s;
  __syncthreads();
  float ss = 0.0f;
#pragma unroll
  for (int w = 0; w < 8; ++w) ss += red[w];
  float inv = 1.0f / ss;
  _Float16* orow = Ah + (size_t)r * SSP;
  for (int i = threadIdx.x; i < SSP; i += 256)
    orow[i] = (_Float16)(__expf(row[i] - mm) * inv);
}

// ---- f_out = FV * FA (complex), packed per-pixel A' [k][b][128] f16 --------
// A'[b, c'] : c'<64 -> Re(f_out[b,c]), c'>=64 -> Im(f_out[b,c]).

__global__ void fout_pack_kernel(const float* __restrict__ FV,
                                 const float* __restrict__ FA,
                                 _Float16* __restrict__ Ap) {
  int gid = blockIdx.x * 256 + threadIdx.x;   // RQK * NLM
  int k = gid & (NLM - 1);
  int r = gid >> 10;                          // b*64 + c
  int b = r >> 6, c = r & 63;
  float vr = FV[(size_t)r * NC2 + 2 * k], vi = FV[(size_t)r * NC2 + 2 * k + 1];
  float ar = FA[(size_t)r * NC2 + 2 * k], ai = FA[(size_t)r * NC2 + 2 * k + 1];
  _Float16* o = Ap + (size_t)k * 4096 + b * 128;
  o[c]      = (_Float16)(vr * ar - vi * ai);
  o[64 + c] = (_Float16)(vr * ai + vi * ar);
}

// ---- per-pixel complex-folded B''^T [k][128][128] (K-major):
// B''T[o][c]=Wr, B''T[o][c+64]=-Wi, B''T[o+64][c]=Wi, B''T[o+64][c+64]=Wr.

__global__ void bpack_kernel(const float* __restrict__ Wspec,
                             _Float16* __restrict__ Bpp) {
  int gid = blockIdx.x * 256 + threadIdx.x;   // NLM * 4096
  int k  = gid >> 12;
  int oc = gid & 4095;
  int o = oc >> 6, c = oc & 63;
  const float* w = Wspec + (size_t)(1536 + o * 64 + c) * NC2 + 2 * k;
  float wr = w[0], wi = w[1];
  _Float16* bb = Bpp + (size_t)k * 16384;
  bb[o * 128 + c]               = (_Float16)wr;
  bb[o * 128 + 64 + c]          = (_Float16)(-wi);
  bb[(64 + o) * 128 + c]        = (_Float16)wi;
  bb[(64 + o) * 128 + 64 + c]   = (_Float16)wr;
}

// ---- final batched per-pixel GEMM: [32 x 128] @ [128 x 128] with scatter ---

__global__ __launch_bounds__(256) void final_kernel(
    const _Float16* __restrict__ Ap, const _Float16* __restrict__ BppT,
    const int* __restrict__ lm_idx, float* __restrict__ out) {
  __shared__ __align__(16) _Float16 sA[32 * 128];    // 8 KB
  __shared__ __align__(16) _Float16 sB[128 * 128];   // 32 KB (K-major B^T)
  const int k = blockIdx.x;
  const int t = threadIdx.x;
  {
    const _Float16* src = Ap + (size_t)k * 4096;
    int r = t >> 3, cb = (t & 7) * 16;
    lds_cp16(&sA[r * 128 + cb],     &src[r * 128 + cb]);
    lds_cp16(&sA[r * 128 + cb + 8], &src[r * 128 + cb + 8]);
  }
  {
    const _Float16* src = BppT + (size_t)k * 16384;
    int base = t * 64;
#pragma unroll
    for (int i = 0; i < 8; ++i)
      lds_cp16(&sB[base + i * 8], &src[base + i * 8]);
  }
  async_drain();
  __syncthreads();
  const int wave = t >> 5;
  const int lane = t & 31;
  const int mt = wave & 1;
  const int nb = (wave >> 1) * 32;
  v8f acc0 = {}, acc1 = {};
#pragma unroll
  for (int kk = 0; kk < 4; ++kk) {
    v16h a  = frag_a(sA, mt * 16, kk * 32, 128, lane);
    v16h b0 = frag_bt(sB, nb, kk * 32, 128, lane);
    acc0 = WMMA_F32_F16(a, b0, acc0);
    v16h b1 = frag_bt(sB, nb + 16, kk * 32, 128, lane);
    acc1 = WMMA_F32_F16(a, b1, acc1);
  }
  const int sp = lm_idx[k];
  const int r0 = mt * 16 + ((lane & 16) >> 1);
#pragma unroll
  for (int j = 0; j < 8; ++j) {
    int b = r0 + j;
    int c0 = nb + (lane & 15);
    int c1 = c0 + 16;
    if (c0 < 64) out[((size_t)(b * 64 + c0) * SSP + sp) * 2]           = acc0[j];
    else         out[((size_t)(b * 64 + c0 - 64) * SSP + sp) * 2 + 1]  = acc0[j];
    if (c1 < 64) out[((size_t)(b * 64 + c1) * SSP + sp) * 2]           = acc1[j];
    else         out[((size_t)(b * 64 + c1 - 64) * SSP + sp) * 2 + 1]  = acc1[j];
  }
}

// ---------------------------------------------------------------------------

extern "C" void kernel_launch(void* const* d_in, const int* in_sizes, int n_in,
                              void* d_out, int out_size, void* d_ws, size_t ws_size,
                              hipStream_t stream) {
  const float* f_in  = (const float*)d_in[0];
  const float* k_q   = (const float*)d_in[1];
  const float* k_k   = (const float*)d_in[2];
  const float* k_v   = (const float*)d_in[3];
  const float* k_o   = (const float*)d_in[4];
  const float* Ymat  = (const float*)d_in[5];   // complex64 interleaved
  const float* Ypinv = (const float*)d_in[6];   // complex64 interleaved
  const int*   lmidx = (const int*)d_in[7];
  float* out = (float*)d_out;                   // complex64 -> interleaved f32

  char* ws = (char*)d_ws;
  size_t off = 0;
  auto carve = [&](size_t bytes) -> void* {
    void* p = ws + off;
    off += (bytes + 255) & ~(size_t)255;
    return p;
  };
  _Float16* YpTT  = (_Float16*)carve((size_t)NC2 * SSP * 2);   // [2048][2016] f16
  _Float16* Ym2TT = (_Float16*)carve((size_t)NC2 * NC2 * 2);   // [2048][2048] f16
  _Float16* Wh    = (_Float16*)carve((size_t)WROWS * SSP * 2); // weights f16
  float*    Wsp   = (float*)carve((size_t)WROWS * NC2 * 4);    // spectral weights
  _Float16* Ph    = (_Float16*)carve((size_t)RQK * NC2 * 2);   // [Pr|Pi] f16
  float*    FVb   = (float*)carve((size_t)RQK * NC2 * 4);      // FV complex f32
  float*    Asp   = (float*)carve((size_t)RQK * NC2 * 4);      // attention logits
  _Float16* Ah    = (_Float16*)carve((size_t)RQK * SSP * 2);   // softmaxed f16
  float*    FAb   = (float*)carve((size_t)RQK * NC2 * 4);      // FA complex f32
  _Float16* Ap    = (_Float16*)carve((size_t)NLM * 4096 * 2);  // per-pixel A'
  _Float16* BppT  = (_Float16*)carve((size_t)NLM * 16384 * 2); // per-pixel B''^T
  (void)ws_size; (void)n_in; (void)in_sizes; (void)out_size;

  const int outF = 2 * BB * CCH * SSP;  // 8,257,536 floats (re/im interleaved)
  zero_f32_kernel<<<(outF + 255) / 256, 256, 0, stream>>>(out, outF);

  pack_yptt_kernel<<<(NC2 * SSP) / 256, 256, 0, stream>>>(Ypinv, YpTT);
  pack_ym2tt_kernel<<<(NC2 * NC2) / 256, 256, 0, stream>>>(Ymat, Ym2TT);

  const int wqkv = 512 * SSP;
  cvt_f16_kernel<<<(wqkv + 255) / 256, 256, 0, stream>>>(k_q, Wh, wqkv);
  cvt_f16_kernel<<<(wqkv + 255) / 256, 256, 0, stream>>>(k_k, Wh + (size_t)512 * SSP, wqkv);
  cvt_f16_kernel<<<(wqkv + 255) / 256, 256, 0, stream>>>(k_v, Wh + (size_t)1024 * SSP, wqkv);
  const int wo = 4096 * SSP;
  cvt_f16_kernel<<<(wo + 255) / 256, 256, 0, stream>>>(k_o, Wh + (size_t)1536 * SSP, wo);

  // 1) spectral weights: [5632,2016] @ YpTT^T -> [5632,2048]
  gemm_wmma_kernel<<<dim3(NC2 / 64, WROWS / 64), 128, 0, stream>>>(
      Wh, YpTT, Wsp, WROWS, NC2, SSP);

  // 2) FQ/FK/FV + P = FQ*conj(FK)
  qkv_kernel<<<(RQK * NLM) / 256, 256, 0, stream>>>(f_in, Wsp, lmidx, Ph, FVb);

  // 3) inverse SHT (real part): [2048,2048] @ Ym2TT^T -> [2048,2048]
  gemm_wmma_kernel<<<dim3(NC2 / 64, RQK / 64), 128, 0, stream>>>(
      Ph, Ym2TT, Asp, RQK, NC2, NC2);

  // 4) softmax over spatial samples
  softmax_kernel<<<RQK, 256, 0, stream>>>(Asp, Ah);

  // 5) forward SHT of attention: [2048,2016] @ YpTT^T -> [2048,2048]
  gemm_wmma_kernel<<<dim3(NC2 / 64, RQK / 64), 128, 0, stream>>>(
      Ah, YpTT, FAb, RQK, NC2, SSP);

  // 6) f_out = FV * FA, pack per-pixel operands
  fout_pack_kernel<<<(RQK * NLM) / 256, 256, 0, stream>>>(FVb, FAb, Ap);
  bpack_kernel<<<(NLM * 4096) / 256, 256, 0, stream>>>(Wsp, BppT);

  // 7) per-pixel complex output projection + scatter into padded grid
  final_kernel<<<NLM, 256, 0, stream>>>(Ap, BppT, lmidx, out);
}
